// vit_gcn_22273700397177
// MI455X (gfx1250) — matrix operand order repaired and checked by reading, not compiled, not run on hardware
//
#include <hip/hip_runtime.h>
#include <hip/hip_bf16.h>
#include <math.h>

// ---------------- problem constants (match reference) ----------------
#define B_GRAPHS 8
#define NPG0     4096
#define EPG      65536
#define E_TOT    (B_GRAPHS * EPG)       // 524288 edges, constant across levels
#define IN_F     384
#define KP1      2458
#define KP2      1475
#define KP3      738
#define EPS_BN   1e-5f
#define NEG_INF  (-3.0e38f)
#define NT       4                      // N-tiles per wave in GEMM

// ---------------- WMMA types (gfx1250, wave32) ----------------
typedef __attribute__((ext_vector_type(16))) __bf16 v16bf;
typedef __attribute__((ext_vector_type(8)))  float  v8f;

static __device__ __forceinline__ __bf16 f2bf(float f) {
  unsigned u = __builtin_bit_cast(unsigned, f);
  unsigned r = u + 0x7FFFu + ((u >> 16) & 1u);   // round-to-nearest-even
  unsigned short h = (unsigned short)(r >> 16);
  return __builtin_bit_cast(__bf16, h);
}

// =====================================================================
// Pack W[K,N] (f32, row-major) into bf16 B-fragment layout:
//   Wp[((kt*tilesN + tn)*32 + lane)*16 + j] = bf16(W[(kt*32 + 16*half + j)*N + tn*16 + r])
// so each lane's 16-element fragment is one contiguous 32-byte load.
// =====================================================================
__global__ void pack_w_bf16(const float* __restrict__ W, __bf16* __restrict__ Wp,
                            int K, int N) {
  long long i = (long long)blockIdx.x * blockDim.x + threadIdx.x;
  if (i >= (long long)K * N) return;
  int j    = (int)(i & 15);
  int lane = (int)((i >> 4) & 31);
  long long tile = i >> 9;          // /512 elements per 32x16 tile
  int tilesN = N >> 4;
  int tn = (int)(tile % tilesN);
  int kt = (int)(tile / tilesN);
  int half = lane >> 4, r = lane & 15;
  int kk = kt * 32 + 16 * half + j;
  Wp[i] = f2bf(W[(long long)kk * N + tn * 16 + r]);
}

// =====================================================================
// Pack A[M,K] (f32, row-major) into bf16 A-fragment layout (one-time RNE
// conversion; padded rows written as zero so GEMM needs no bounds checks):
//   Ap[((tm*(K/32) + kt)*32 + lane)*16 + j] =
//       bf16(A[(tm*16 + r)*K + kt*32 + 16*half + j])
// =====================================================================
__global__ void pack_a_bf16(const float* __restrict__ A, __bf16* __restrict__ Ap,
                            int M, int K, int tilesM) {
  long long i = (long long)blockIdx.x * blockDim.x + threadIdx.x;
  long long total = (long long)tilesM * 16 * K;
  if (i >= total) return;
  int j    = (int)(i & 15);
  int lane = (int)((i >> 4) & 31);
  long long rest = i >> 9;          // /512 elements per 16x32 tile
  int ktiles = K >> 5;
  int kt = (int)(rest % ktiles);
  int tm = (int)(rest / ktiles);
  int half = lane >> 4, r = lane & 15;
  int row = tm * 16 + r;
  int ka = kt * 32 + 16 * half + j;
  float v = (row < M) ? A[(long long)row * K + ka] : 0.0f;
  Ap[i] = f2bf(v);
}

// =====================================================================
// GEMM: Y[16*tilesM,N] = A * W with both operands pre-packed as bf16
// fragments. One wave32 computes a 16x64 strip (4 N-tiles): per K-step
// one 32B A-fragment load + four 32B B-fragment loads + 4x
// v_wmma_f32_16x16x32_bf16 -- no conversion VALU, no bounds checks.
// D layout: elem v -> row M = v + 8*half, col N = lane&15.
// =====================================================================
__global__ void gemm_wmma_bf16(const __bf16* __restrict__ Ap,
                               const __bf16* __restrict__ Bp,
                               float* __restrict__ Y,
                               int tilesM, int K, int N) {
  int wave   = (int)((blockIdx.x * blockDim.x + threadIdx.x) >> 5);
  int lane   = threadIdx.x & 31;
  int tilesN = N >> 4;
  int strips = tilesN / NT;
  if (wave >= tilesM * strips) return;            // wave-uniform exit
  int tm = wave / strips;
  int tn0 = (wave % strips) * NT;
  int half = lane >> 4, r = lane & 15;
  int ktiles = K >> 5;

  const v16bf* ap = (const v16bf*)(Ap + ((long long)tm * ktiles * 32 + lane) * 16);
  const v16bf* bp = (const v16bf*)(Bp + ((long long)tn0 * 32 + lane) * 16);
  const long long bstep = (long long)tilesN * 32;   // v16bf units per K-tile

  v8f acc0 = {}, acc1 = {}, acc2 = {}, acc3 = {};
#pragma unroll 2
  for (int kt = 0; kt < ktiles; ++kt) {
    __builtin_prefetch(ap + 64, 0, 1);              // A two K-tiles ahead
    v16bf a = ap[0];
    // consecutive N-tiles are 512 bf16 = 32 v16bf apart
    acc0 = __builtin_amdgcn_wmma_f32_16x16x32_bf16(false, a, false, bp[0],  (short)0, acc0, false, false);
    acc1 = __builtin_amdgcn_wmma_f32_16x16x32_bf16(false, a, false, bp[32], (short)0, acc1, false, false);
    acc2 = __builtin_amdgcn_wmma_f32_16x16x32_bf16(false, a, false, bp[64], (short)0, acc2, false, false);
    acc3 = __builtin_amdgcn_wmma_f32_16x16x32_bf16(false, a, false, bp[96], (short)0, acc3, false, false);
    ap += 32;
    bp += bstep;
  }
#pragma unroll
  for (int v = 0; v < 8; ++v) {
    float* yr = Y + (long long)(tm * 16 + v + 8 * half) * N + tn0 * 16 + r;
    yr[0]  = acc0[v];
    yr[16] = acc1[v];
    yr[32] = acc2[v];
    yr[48] = acc3[v];
  }
}

// ---------------- utility fills ----------------
__global__ void fill_f32(float* p, float v, long long n) {
  long long i = (long long)blockIdx.x * blockDim.x + threadIdx.x;
  if (i < n) p[i] = v;
}
__global__ void fill_i32(int* p, int v, long long n) {
  long long i = (long long)blockIdx.x * blockDim.x + threadIdx.x;
  if (i < n) p[i] = v;
}
__global__ void bias_init(float* out, const float* __restrict__ b, int n, int F) {
  long long i = (long long)blockIdx.x * blockDim.x + threadIdx.x;
  if (i < (long long)n * F) out[i] = b[i % F];
}

// ---------------- GCN normalization ----------------
__global__ void edge_deg(const int* __restrict__ dst, const float* __restrict__ em,
                         float* __restrict__ deg) {
  int e = blockIdx.x * blockDim.x + threadIdx.x;
  if (e < E_TOT) {
    float w = em[e];
    if (w != 0.0f) atomicAdd(&deg[dst[e]], w);
  }
}
__global__ void deg_to_dis(float* deg, int n) {
  int i = blockIdx.x * blockDim.x + threadIdx.x;
  if (i < n) { float d = deg[i]; deg[i] = (d > 0.0f) ? rsqrtf(fmaxf(d, 1e-12f)) : 0.0f; }
}
__global__ void edge_coef(const int* __restrict__ src, const int* __restrict__ dst,
                          const float* __restrict__ em, const float* __restrict__ dis,
                          float* __restrict__ coef) {
  int e = blockIdx.x * blockDim.x + threadIdx.x;
  if (e < E_TOT) coef[e] = dis[src[e]] * dis[dst[e]] * em[e];
}
// out[dst] += h[src] * coef ; 4 features per thread (float4 gather + 4 atomics)
__global__ void edge_agg4(const int* __restrict__ src, const int* __restrict__ dst,
                          const float* __restrict__ coef, const float* __restrict__ h,
                          float* __restrict__ out, int F) {
  int F4 = F >> 2;
  long long idx = (long long)blockIdx.x * blockDim.x + threadIdx.x;
  if (idx >= (long long)E_TOT * F4) return;
  int e  = (int)(idx / F4);
  int c4 = (int)(idx % F4) << 2;
  float cf = coef[e];
  if (cf == 0.0f) return;
  float4 hv = *(const float4*)(h + (long long)src[e] * F + c4);
  float* o = out + (long long)dst[e] * F + c4;
  atomicAdd(o + 0, hv.x * cf);
  atomicAdd(o + 1, hv.y * cf);
  atomicAdd(o + 2, hv.z * cf);
  atomicAdd(o + 3, hv.w * cf);
}

// ---------------- BatchNorm (biased var) + ReLU ----------------
__global__ void bn_stats(const float* __restrict__ x, int n, int F,
                         float* __restrict__ mean, float* __restrict__ istd) {
  __shared__ float s1[256], s2[256];
  int c = blockIdx.x;
  float sum = 0.f, sq = 0.f;
  for (int r = threadIdx.x; r < n; r += blockDim.x) {
    float v = x[(long long)r * F + c]; sum += v; sq += v * v;
  }
  s1[threadIdx.x] = sum; s2[threadIdx.x] = sq; __syncthreads();
  for (int o = 128; o > 0; o >>= 1) {
    if ((int)threadIdx.x < o) { s1[threadIdx.x] += s1[threadIdx.x + o]; s2[threadIdx.x] += s2[threadIdx.x + o]; }
    __syncthreads();
  }
  if (threadIdx.x == 0) {
    float mu = s1[0] / (float)n;
    float var = s2[0] / (float)n - mu * mu;
    mean[c] = mu; istd[c] = rsqrtf(var + EPS_BN);
  }
}
__global__ void bn_apply_relu(float* x, const float* __restrict__ mean,
                              const float* __restrict__ istd,
                              const float* __restrict__ g, const float* __restrict__ be,
                              int n, int F) {
  long long i = (long long)blockIdx.x * blockDim.x + threadIdx.x;
  if (i >= (long long)n * F) return;
  int c = (int)(i % F);
  float v = (x[i] - mean[c]) * istd[c] * g[c] + be[c];
  x[i] = fmaxf(v, 0.0f);
}

// ---------------- SAGPool score: GraphConv -> scalar ----------------
__global__ void node_scores(const float* __restrict__ x,
                            const float* __restrict__ wrel,
                            const float* __restrict__ wroot,
                            const float* __restrict__ brel,
                            float* __restrict__ p, float* __restrict__ score,
                            int n, int F) {
  int i = blockIdx.x * blockDim.x + threadIdx.x;
  if (i >= n) return;
  const float* row = x + (long long)i * F;
  float a = 0.f, b = 0.f;
  for (int f = 0; f < F; f += 4) {
    float4 v = *(const float4*)(row + f);
    float4 wr = *(const float4*)(wrel + f);
    float4 wo = *(const float4*)(wroot + f);
    a += v.x * wr.x + v.y * wr.y + v.z * wr.z + v.w * wr.w;
    b += v.x * wo.x + v.y * wo.y + v.z * wo.z + v.w * wo.w;
  }
  p[i] = a; score[i] = b + brel[0];
}
__global__ void edge_score(const int* __restrict__ src, const int* __restrict__ dst,
                           const float* __restrict__ em, const float* __restrict__ p,
                           float* __restrict__ score) {
  int e = blockIdx.x * blockDim.x + threadIdx.x;
  if (e < E_TOT) {
    float w = em[e];
    if (w != 0.0f) atomicAdd(&score[dst[e]], w * p[src[e]]);
  }
}

// ---------------- per-graph top-k via LDS bitonic sort ----------------
__global__ void topk_kernel(const float* __restrict__ score, int npg, int k, int pw,
                            int* __restrict__ perm, float* __restrict__ tanhv) {
  __shared__ float sv[4096];
  __shared__ int   si[4096];
  int g = blockIdx.x;
  for (int i = threadIdx.x; i < pw; i += blockDim.x) {
    sv[i] = (i < npg) ? score[g * npg + i] : NEG_INF;
    si[i] = i;
  }
  __syncthreads();
  for (int size = 2; size <= pw; size <<= 1) {
    for (int stride = size >> 1; stride > 0; stride >>= 1) {
      for (int i = threadIdx.x; i < pw; i += blockDim.x) {
        int j = i ^ stride;
        if (j > i) {
          bool desc = ((i & size) == 0);
          float vi = sv[i], vj = sv[j];
          bool sw = desc ? (vi < vj) : (vi > vj);
          if (sw) { sv[i] = vj; sv[j] = vi; int t = si[i]; si[i] = si[j]; si[j] = t; }
        }
      }
      __syncthreads();
    }
  }
  for (int i = threadIdx.x; i < k; i += blockDim.x) {
    perm[g * k + i]  = g * npg + si[i];
    tanhv[g * k + i] = tanhf(sv[i]);
  }
}
__global__ void pool_gather(const float* __restrict__ x, const int* __restrict__ perm,
                            const float* __restrict__ tanhv, float* __restrict__ xn,
                            int nk, int F) {
  int F4 = F >> 2;
  long long i = (long long)blockIdx.x * blockDim.x + threadIdx.x;
  if (i >= (long long)nk * F4) return;
  int node = (int)(i / F4), c4 = (int)(i % F4) << 2;
  float t = tanhv[node];
  float4 v = *(const float4*)(x + (long long)perm[node] * F + c4);
  float4 o; o.x = v.x * t; o.y = v.y * t; o.z = v.z * t; o.w = v.w * t;
  *(float4*)(xn + (long long)node * F + c4) = o;
}
__global__ void set_inv(const int* __restrict__ perm, int* __restrict__ inv, int nk) {
  int i = blockIdx.x * blockDim.x + threadIdx.x;
  if (i < nk) inv[perm[i]] = i;
}
__global__ void remap_edges(const int* __restrict__ src, const int* __restrict__ dst,
                            const float* __restrict__ em, const int* __restrict__ inv,
                            int* __restrict__ src2, int* __restrict__ dst2,
                            float* __restrict__ em2) {
  int e = blockIdx.x * blockDim.x + threadIdx.x;
  if (e >= E_TOT) return;
  int ns = inv[src[e]], nd = inv[dst[e]];
  bool keep = (ns >= 0) && (nd >= 0);
  src2[e] = keep ? ns : 0;
  dst2[e] = keep ? nd : 0;
  em2[e]  = keep ? em[e] : 0.0f;
}

// ---------------- global mean + max readout ----------------
__global__ void mean_max_kernel(const float* __restrict__ h, float* __restrict__ out) {
  int idx = blockIdx.x * blockDim.x + threadIdx.x;
  if (idx >= B_GRAPHS * 192) return;
  int g = idx / 192, f = idx % 192;
  const float* p = h + (long long)g * KP3 * 192 + f;
  float s = 0.f, m = NEG_INF;
  for (int j = 0; j < KP3; ++j) { float v = p[(long long)j * 192]; s += v; m = fmaxf(m, v); }
  out[idx] = s / (float)KP3 + m;
}

// =====================================================================
// Host-side orchestration
// =====================================================================
static inline unsigned gridFor(long long n, int bs) { return (unsigned)((n + bs - 1) / bs); }

static void run_gcn_conv(const float* xin, const float* W, const float* bias,
                         int n, int Fin, int Fout,
                         const int* src, const int* dst, const float* em,
                         float* h_tmp, float* outbuf, float* deg, float* coef,
                         __bf16* apack, __bf16* wpack, hipStream_t s) {
  // pack A and W into bf16 fragments (one-time RNE conversion, halves GEMM traffic)
  int tilesM = (n + 15) / 16;
  pack_a_bf16<<<gridFor((long long)tilesM * 16 * Fin, 256), 256, 0, s>>>(xin, apack, n, Fin, tilesM);
  pack_w_bf16<<<gridFor((long long)Fin * Fout, 256), 256, 0, s>>>(W, wpack, Fin, Fout);
  // h = x @ W via WMMA
  long long waves = (long long)tilesM * (Fout / 16 / NT);
  gemm_wmma_bf16<<<gridFor(waves * 32, 256), 256, 0, s>>>(apack, wpack, h_tmp, tilesM, Fin, Fout);
  // degree -> D^-1/2
  fill_f32<<<gridFor(n, 256), 256, 0, s>>>(deg, 0.0f, n);
  edge_deg<<<gridFor(E_TOT, 256), 256, 0, s>>>(dst, em, deg);
  deg_to_dis<<<gridFor(n, 256), 256, 0, s>>>(deg, n);
  edge_coef<<<gridFor(E_TOT, 256), 256, 0, s>>>(src, dst, em, deg, coef);
  // out = bias ; out[dst] += h[src] * coef
  bias_init<<<gridFor((long long)n * Fout, 256), 256, 0, s>>>(outbuf, bias, n, Fout);
  edge_agg4<<<gridFor((long long)E_TOT * (Fout >> 2), 256), 256, 0, s>>>(src, dst, coef, h_tmp, outbuf, Fout);
}

extern "C" void kernel_launch(void* const* d_in, const int* in_sizes, int n_in,
                              void* d_out, int out_size, void* d_ws, size_t ws_size,
                              hipStream_t stream) {
  (void)in_sizes; (void)n_in; (void)out_size; (void)ws_size;
  // -------- inputs (setup_inputs dict order) --------
  const float* x   = (const float*)d_in[0];
  const int*   ei  = (const int*)d_in[1];
  const float* W1  = (const float*)d_in[2];  const float* b1 = (const float*)d_in[3];
  const float* W2  = (const float*)d_in[4];  const float* b2 = (const float*)d_in[5];
  const float* W3  = (const float*)d_in[6];  const float* b3 = (const float*)d_in[7];
  const float* W4  = (const float*)d_in[8];  const float* b4 = (const float*)d_in[9];
  const float* g1  = (const float*)d_in[10]; const float* be1 = (const float*)d_in[11];
  const float* g2  = (const float*)d_in[12]; const float* be2 = (const float*)d_in[13];
  const float* g3  = (const float*)d_in[14]; const float* be3 = (const float*)d_in[15];
  const float* p1_wrel = (const float*)d_in[16]; const float* p1_brel = (const float*)d_in[17];
  const float* p1_wroot = (const float*)d_in[18];
  const float* p2_wrel = (const float*)d_in[19]; const float* p2_brel = (const float*)d_in[20];
  const float* p2_wroot = (const float*)d_in[21];
  const float* p3_wrel = (const float*)d_in[22]; const float* p3_brel = (const float*)d_in[23];
  const float* p3_wroot = (const float*)d_in[24];
  float* out = (float*)d_out;

  const int* src0 = ei;
  const int* dst0 = ei + E_TOT;

  // -------- bump allocation from workspace --------
  char* ws = (char*)d_ws;
  size_t off = 0;
  auto alloc = [&](size_t bytes) -> char* {
    char* p = ws + off;
    off += (bytes + 255) & ~(size_t)255;
    return p;
  };
  const int n1 = B_GRAPHS * NPG0;   // 32768 (mult of 16)
  const int n2 = B_GRAPHS * KP1;    // 19664 (mult of 16)
  const int n3 = B_GRAPHS * KP2;    // 11800 (pack_a zero-pads to 11808)
  const int n4 = B_GRAPHS * KP3;    // 5904  (mult of 16)

  float* buf1 = (float*)alloc((size_t)n1 * 512 * 4);   // gemm h temp (padded rows ok)
  float* buf2 = (float*)alloc((size_t)n1 * 512 * 4);   // agg output / bn in-place
  float* buf3 = (float*)alloc((size_t)n2 * 512 * 4);   // pooled features
  float* em0  = (float*)alloc((size_t)E_TOT * 4);
  int*   srcA = (int*)alloc((size_t)E_TOT * 4);
  int*   dstA = (int*)alloc((size_t)E_TOT * 4);
  float* emA  = (float*)alloc((size_t)E_TOT * 4);
  int*   srcB = (int*)alloc((size_t)E_TOT * 4);
  int*   dstB = (int*)alloc((size_t)E_TOT * 4);
  float* emB  = (float*)alloc((size_t)E_TOT * 4);
  float* coef = (float*)alloc((size_t)E_TOT * 4);
  float* deg  = (float*)alloc((size_t)n1 * 4);
  float* scor = (float*)alloc((size_t)n1 * 4);
  float* pbuf = (float*)alloc((size_t)n1 * 4);
  float* mean = (float*)alloc(512 * 4);
  float* istd = (float*)alloc(512 * 4);
  int*   perm = (int*)alloc((size_t)n2 * 4);
  float* tanv = (float*)alloc((size_t)n2 * 4);
  int*   invmap = (int*)alloc((size_t)n1 * 4);
  __bf16* apack = (__bf16*)alloc((size_t)n1 * 512 * 2);  // packed A (max 32768x512)
  __bf16* wpack = (__bf16*)alloc((size_t)512 * 512 * 2); // packed W (max 512x512)

  fill_f32<<<gridFor(E_TOT, 256), 256, 0, stream>>>(em0, 1.0f, E_TOT);

  // ================= Level 1 =================
  run_gcn_conv(x, W1, b1, n1, IN_F, 512, src0, dst0, em0, buf1, buf2, deg, coef, apack, wpack, stream);
  bn_stats<<<512, 256, 0, stream>>>(buf2, n1, 512, mean, istd);
  bn_apply_relu<<<gridFor((long long)n1 * 512, 256), 256, 0, stream>>>(buf2, mean, istd, g1, be1, n1, 512);
  node_scores<<<gridFor(n1, 128), 128, 0, stream>>>(buf2, p1_wrel, p1_wroot, p1_brel, pbuf, scor, n1, 512);
  edge_score<<<gridFor(E_TOT, 256), 256, 0, stream>>>(src0, dst0, em0, pbuf, scor);
  topk_kernel<<<B_GRAPHS, 256, 0, stream>>>(scor, NPG0, KP1, 4096, perm, tanv);
  pool_gather<<<gridFor((long long)n2 * 128, 256), 256, 0, stream>>>(buf2, perm, tanv, buf3, n2, 512);
  fill_i32<<<gridFor(n1, 256), 256, 0, stream>>>(invmap, -1, n1);
  set_inv<<<gridFor(n2, 256), 256, 0, stream>>>(perm, invmap, n2);
  remap_edges<<<gridFor(E_TOT, 256), 256, 0, stream>>>(src0, dst0, em0, invmap, srcA, dstA, emA);

  // ================= Level 2 =================
  run_gcn_conv(buf3, W2, b2, n2, 512, 512, srcA, dstA, emA, buf1, buf2, deg, coef, apack, wpack, stream);
  bn_stats<<<512, 256, 0, stream>>>(buf2, n2, 512, mean, istd);
  bn_apply_relu<<<gridFor((long long)n2 * 512, 256), 256, 0, stream>>>(buf2, mean, istd, g2, be2, n2, 512);
  node_scores<<<gridFor(n2, 128), 128, 0, stream>>>(buf2, p2_wrel, p2_wroot, p2_brel, pbuf, scor, n2, 512);
  edge_score<<<gridFor(E_TOT, 256), 256, 0, stream>>>(srcA, dstA, emA, pbuf, scor);
  topk_kernel<<<B_GRAPHS, 256, 0, stream>>>(scor, KP1, KP2, 4096, perm, tanv);
  pool_gather<<<gridFor((long long)n3 * 128, 256), 256, 0, stream>>>(buf2, perm, tanv, buf3, n3, 512);
  fill_i32<<<gridFor(n2, 256), 256, 0, stream>>>(invmap, -1, n2);
  set_inv<<<gridFor(n3, 256), 256, 0, stream>>>(perm, invmap, n3);
  remap_edges<<<gridFor(E_TOT, 256), 256, 0, stream>>>(srcA, dstA, emA, invmap, srcB, dstB, emB);

  // ================= Level 3 =================
  run_gcn_conv(buf3, W3, b3, n3, 512, 256, srcB, dstB, emB, buf1, buf2, deg, coef, apack, wpack, stream);
  bn_stats<<<256, 256, 0, stream>>>(buf2, n3, 256, mean, istd);
  bn_apply_relu<<<gridFor((long long)n3 * 256, 256), 256, 0, stream>>>(buf2, mean, istd, g3, be3, n3, 256);
  node_scores<<<gridFor(n3, 128), 128, 0, stream>>>(buf2, p3_wrel, p3_wroot, p3_brel, pbuf, scor, n3, 256);
  edge_score<<<gridFor(E_TOT, 256), 256, 0, stream>>>(srcB, dstB, emB, pbuf, scor);
  topk_kernel<<<B_GRAPHS, 256, 0, stream>>>(scor, KP2, KP3, 2048, perm, tanv);
  pool_gather<<<gridFor((long long)n4 * 64, 256), 256, 0, stream>>>(buf2, perm, tanv, buf3, n4, 256);
  fill_i32<<<gridFor(n3, 256), 256, 0, stream>>>(invmap, -1, n3);
  set_inv<<<gridFor(n4, 256), 256, 0, stream>>>(perm, invmap, n4);
  remap_edges<<<gridFor(E_TOT, 256), 256, 0, stream>>>(srcB, dstB, emB, invmap, srcA, dstA, emA);

  // ================= Final GCN + readout =================
  run_gcn_conv(buf3, W4, b4, n4, 256, 192, srcA, dstA, emA, buf1, buf2, deg, coef, apack, wpack, stream);
  mean_max_kernel<<<gridFor(B_GRAPHS * 192, 256), 256, 0, stream>>>(buf2, out);
}
